// EncoderLayer_13503377179139
// MI455X (gfx1250) — matrix operand order repaired and checked by reading, not compiled, not run on hardware
//
#include <hip/hip_runtime.h>

typedef unsigned short u16;
typedef __bf16 bf16t;
typedef __attribute__((ext_vector_type(16))) bf16t v16bf;
typedef __attribute__((ext_vector_type(8)))  float v8f;

union FragU { uint4 q[2]; v16bf v; };

__device__ __forceinline__ u16 f2bf(float f) {
  union { float f; unsigned u; } x; x.f = f;
  unsigned r = x.u + 0x7FFFu + ((x.u >> 16) & 1u);
  return (u16)(r >> 16);
}
__device__ __forceinline__ v8f zero8() {
  v8f z = {0.f,0.f,0.f,0.f,0.f,0.f,0.f,0.f}; return z;
}

// ---------------------------------------------------------------- convert x
__global__ __launch_bounds__(256) void conv_bf16_k(const float* __restrict__ in,
                                                   u16* __restrict__ out, int n) {
  int i = blockIdx.x * 256 + threadIdx.x;
  if (i < n) out[i] = f2bf(in[i]);
}

// ------------------------------------------------- repack weight -> bf16 CM
// in : [H][D][Kd] f32 ; out: [H*Kd][D] bf16  (column-major B for the GEMM)
__global__ __launch_bounds__(256) void repack_w_k(const float* __restrict__ in,
                                                  u16* __restrict__ out,
                                                  int H, int D, int Kd) {
  int idx = blockIdx.x * 256 + threadIdx.x;
  int total = H * D * Kd;
  if (idx >= total) return;
  int n = idx / D;
  int d = idx - n * D;
  int h = n / Kd;
  int k = n - h * Kd;
  out[idx] = f2bf(in[((size_t)h * D + d) * (size_t)Kd + k]);
}

// ---------------------------------------------------------------- GEMM bf16
// C = A[M,K](row) * Bcm[N,K](row == B column-major) ; EPI:
//   0: bf16 out in head layout [(b*16+h)*2048+s][64] (+bias)   (QKV)
//   1: f32 out = acc + bias + res                               (proj / FF2)
//   2: bf16 out = relu(acc + bias)                              (FF1)
template<int EPI>
__global__ __launch_bounds__(256) void gemm_k(
    const u16* __restrict__ A, const u16* __restrict__ B,
    const float* __restrict__ bias, const float* __restrict__ res,
    float* __restrict__ outF, u16* __restrict__ outB,
    int M, int N, int K) {
  __shared__ u16 As[128 * 40];
  __shared__ u16 Bs[64 * 40];
  const int tid  = threadIdx.x;
  const int m0   = blockIdx.y * 128;
  const int n0   = blockIdx.x * 64;
  const int wave = tid >> 5, lane = tid & 31;
  const int wm = wave & 3, wn = wave >> 2;
  const int l15 = lane & 15, lh = lane >> 4;

  v8f acc[2][2];
  acc[0][0] = zero8(); acc[0][1] = zero8();
  acc[1][0] = zero8(); acc[1][1] = zero8();

  const int ar = tid >> 1, ac = (tid & 1) << 4;
  const int bn = tid >> 2, bseg = (tid & 3) << 3;
  const u16* Ap = A + (size_t)(m0 + ar) * K + ac;
  const u16* Bp = B + (size_t)(n0 + bn) * K + bseg;

  for (int k0 = 0; k0 < K; k0 += 32) {
    uint4 a0 = *(const uint4*)(Ap + k0);
    uint4 a1 = *(const uint4*)(Ap + k0 + 8);
    uint4 b0 = *(const uint4*)(Bp + k0);
    if (k0 + 32 < K) {
      __builtin_prefetch(Ap + k0 + 32, 0, 1);
      __builtin_prefetch(Bp + k0 + 32, 0, 1);
    }
    __syncthreads();
    *(uint4*)(&As[ar * 40 + ac])     = a0;
    *(uint4*)(&As[ar * 40 + ac + 8]) = a1;
    *(uint4*)(&Bs[bn * 40 + bseg])   = b0;
    __syncthreads();

    FragU af[2], bfr[2];
#pragma unroll
    for (int i = 0; i < 2; ++i) {
      int r = (wm * 32 + i * 16 + l15) * 40;
      af[i].q[0] = *(const uint4*)(&As[r + lh * 8]);
      af[i].q[1] = *(const uint4*)(&As[r + 16 + lh * 8]);
    }
#pragma unroll
    for (int j = 0; j < 2; ++j) {
      int r = (wn * 32 + j * 16 + l15) * 40 + lh * 8;
      bfr[j].q[0] = *(const uint4*)(&Bs[r]);
      bfr[j].q[1] = *(const uint4*)(&Bs[r + 16]);
    }
#pragma unroll
    for (int i = 0; i < 2; ++i)
#pragma unroll
      for (int j = 0; j < 2; ++j)
        acc[i][j] = __builtin_amdgcn_wmma_f32_16x16x32_bf16(
            false, af[i].v, false, bfr[j].v, (short)0, acc[i][j], false, false);
  }

#pragma unroll
  for (int i = 0; i < 2; ++i)
#pragma unroll
    for (int j = 0; j < 2; ++j)
#pragma unroll
      for (int v = 0; v < 8; ++v) {
        int mg = m0 + wm * 32 + i * 16 + lh * 8 + v;
        int ng = n0 + wn * 32 + j * 16 + l15;
        float val = acc[i][j][v] + bias[ng];
        if (EPI == 0) {
          int h = ng >> 6, kk = ng & 63, bb = mg >> 11, ss = mg & 2047;
          outB[(((size_t)(bb * 16 + h) * 2048 + ss) << 6) + kk] = f2bf(val);
        } else if (EPI == 1) {
          size_t o = (size_t)mg * N + ng;
          outF[o] = val + res[o];
        } else {
          outB[(size_t)mg * N + ng] = f2bf(fmaxf(val, 0.f));
        }
      }
}

// ------------------------------------------------------- flash attention
// Q,K,V: bf16 [B*H][2048][64]; ctx out: bf16 [B*2048][1024] (heads concat)
__global__ __launch_bounds__(128) void attn_k(
    const u16* __restrict__ Q, const u16* __restrict__ Km,
    const u16* __restrict__ Vm, const int* __restrict__ mask,
    u16* __restrict__ ctx) {
  __shared__ u16 Qs[64 * 72];
  __shared__ u16 Ks[64 * 72];
  __shared__ u16 Vt[64 * 72];
  __shared__ u16 Ps[4 * 16 * 72];
  const int tid = threadIdx.x;
  const int bh = blockIdx.x >> 5;     // b*16+h
  const int qb = blockIdx.x & 31;
  const int b = bh >> 4, h = bh & 15;
  const int wave = tid >> 5, lane = tid & 31;
  const int l15 = lane & 15, lh = lane >> 4;
  const size_t base = (size_t)bh * 2048 * 64;
  const int qbase = qb * 64;

  {  // stage Q tile 64x64
    int r = tid >> 1, c = (tid & 1) * 32;
    const uint4* s = (const uint4*)(Q + base + (size_t)(qbase + r) * 64 + c);
#pragma unroll
    for (int i = 0; i < 4; ++i) *(uint4*)(&Qs[r * 72 + c + i * 8]) = s[i];
  }
  __syncthreads();
  FragU qf[2];
  {
    int qr = (wave * 16 + l15) * 72;
#pragma unroll
    for (int s = 0; s < 2; ++s) {
      qf[s].q[0] = *(const uint4*)(&Qs[qr + s * 32 + lh * 8]);
      qf[s].q[1] = *(const uint4*)(&Qs[qr + s * 32 + 16 + lh * 8]);
    }
  }

  v8f o[4]; o[0] = zero8(); o[1] = zero8(); o[2] = zero8(); o[3] = zero8();
  float oldm[8], lsum[8];
#pragma unroll
  for (int v = 0; v < 8; ++v) { oldm[v] = -3e38f; lsum[v] = 0.f; }

  for (int kv = 0; kv < 2048; kv += 64) {
    __syncthreads();
    {  // stage K tile + V transposed
      int r = tid >> 1, c = (tid & 1) * 32;
      const uint4* ks = (const uint4*)(Km + base + (size_t)(kv + r) * 64 + c);
      const uint4* vs = (const uint4*)(Vm + base + (size_t)(kv + r) * 64 + c);
#pragma unroll
      for (int i = 0; i < 4; ++i) *(uint4*)(&Ks[r * 72 + c + i * 8]) = ks[i];
#pragma unroll
      for (int i = 0; i < 4; ++i) {
        uint4 t = vs[i];
        const u16* tu = (const u16*)&t;
#pragma unroll
        for (int e = 0; e < 8; ++e) Vt[(c + i * 8 + e) * 72 + r] = tu[e];
      }
    }
    __syncthreads();

    // scores = Q * K^T
    v8f sc[4]; sc[0] = zero8(); sc[1] = zero8(); sc[2] = zero8(); sc[3] = zero8();
#pragma unroll
    for (int s = 0; s < 2; ++s)
#pragma unroll
      for (int j = 0; j < 4; ++j) {
        FragU kf;
        int r = (j * 16 + l15) * 72 + s * 32 + lh * 8;
        kf.q[0] = *(const uint4*)(&Ks[r]);
        kf.q[1] = *(const uint4*)(&Ks[r + 16]);
        sc[j] = __builtin_amdgcn_wmma_f32_16x16x32_bf16(
            false, qf[s].v, false, kf.v, (short)0, sc[j], false, false);
      }
    // scale + mask
#pragma unroll
    for (int j = 0; j < 4; ++j) {
      int mk = mask[b * 2048 + kv + j * 16 + l15];
      float mb = mk ? 0.f : -1e9f;
#pragma unroll
      for (int v = 0; v < 8; ++v) sc[j][v] = sc[j][v] * 0.125f + mb;
    }
    // online softmax (row = lh*8 + v inside this wave's 16 rows)
    float rmx[8];
#pragma unroll
    for (int v = 0; v < 8; ++v)
      rmx[v] = fmaxf(fmaxf(sc[0][v], sc[1][v]), fmaxf(sc[2][v], sc[3][v]));
#pragma unroll
    for (int d = 1; d < 16; d <<= 1)
#pragma unroll
      for (int v = 0; v < 8; ++v)
        rmx[v] = fmaxf(rmx[v], __shfl_xor(rmx[v], d, 16));
    float esc[8];
#pragma unroll
    for (int v = 0; v < 8; ++v) {
      float nm = fmaxf(oldm[v], rmx[v]);
      esc[v] = __expf(oldm[v] - nm);
      oldm[v] = nm;
    }
    float rs[8];
#pragma unroll
    for (int v = 0; v < 8; ++v) rs[v] = 0.f;
#pragma unroll
    for (int j = 0; j < 4; ++j)
#pragma unroll
      for (int v = 0; v < 8; ++v) {
        float p = __expf(sc[j][v] - oldm[v]);
        sc[j][v] = p;
        rs[v] += p;
      }
#pragma unroll
    for (int d = 1; d < 16; d <<= 1)
#pragma unroll
      for (int v = 0; v < 8; ++v) rs[v] += __shfl_xor(rs[v], d, 16);
#pragma unroll
    for (int v = 0; v < 8; ++v) lsum[v] = lsum[v] * esc[v] + rs[v];
#pragma unroll
    for (int j = 0; j < 4; ++j)
#pragma unroll
      for (int v = 0; v < 8; ++v) o[j][v] *= esc[v];
    // write P (bf16) to this wave's LDS strip, A-fragment re-layout
#pragma unroll
    for (int j = 0; j < 4; ++j)
#pragma unroll
      for (int v = 0; v < 8; ++v)
        Ps[(wave * 16 + lh * 8 + v) * 72 + j * 16 + l15] = f2bf(sc[j][v]);
    // ctx += P * V
    int pr = (wave * 16 + l15) * 72;
#pragma unroll
    for (int s = 0; s < 2; ++s) {
      FragU pf;
      pf.q[0] = *(const uint4*)(&Ps[pr + s * 32 + lh * 8]);
      pf.q[1] = *(const uint4*)(&Ps[pr + s * 32 + 16 + lh * 8]);
#pragma unroll
      for (int j = 0; j < 4; ++j) {
        FragU vf;
        int r = (j * 16 + l15) * 72 + s * 32 + lh * 8;
        vf.q[0] = *(const uint4*)(&Vt[r]);
        vf.q[1] = *(const uint4*)(&Vt[r + 16]);
        o[j] = __builtin_amdgcn_wmma_f32_16x16x32_bf16(
            false, pf.v, false, vf.v, (short)0, o[j], false, false);
      }
    }
  }

#pragma unroll
  for (int j = 0; j < 4; ++j)
#pragma unroll
    for (int v = 0; v < 8; ++v) {
      float val = o[j][v] / lsum[v];
      int q = qbase + wave * 16 + lh * 8 + v;
      int c = h * 64 + j * 16 + l15;
      ctx[(size_t)(b * 2048 + q) * 1024 + c] = f2bf(val);
    }
}

// ------------------------------------------------------------- layernorm
__global__ __launch_bounds__(256) void ln_k(const float* __restrict__ x,
                                            const float* __restrict__ g,
                                            const float* __restrict__ be,
                                            float* __restrict__ outF,
                                            u16* __restrict__ outB) {
  __shared__ float red[8];
  const int row = blockIdx.x, tid = threadIdx.x;
  const float4 v = ((const float4*)(x + (size_t)row * 1024))[tid];
  float s = v.x + v.y + v.z + v.w;
#pragma unroll
  for (int d = 16; d >= 1; d >>= 1) s += __shfl_xor(s, d, 32);
  if ((tid & 31) == 0) red[tid >> 5] = s;
  __syncthreads();
  float mu = 0.f;
#pragma unroll
  for (int i = 0; i < 8; ++i) mu += red[i];
  mu *= (1.f / 1024.f);
  __syncthreads();
  float d0 = v.x - mu, d1 = v.y - mu, d2 = v.z - mu, d3 = v.w - mu;
  float q = d0 * d0 + d1 * d1 + d2 * d2 + d3 * d3;
#pragma unroll
  for (int d = 16; d >= 1; d >>= 1) q += __shfl_xor(q, d, 32);
  if ((tid & 31) == 0) red[tid >> 5] = q;
  __syncthreads();
  float var = 0.f;
#pragma unroll
  for (int i = 0; i < 8; ++i) var += red[i];
  var *= (1.f / 1024.f);
  const float rstd = rsqrtf(var + 1e-5f);
  const int c = tid * 4;
  float xv[4] = {v.x, v.y, v.z, v.w};
#pragma unroll
  for (int e = 0; e < 4; ++e) {
    float y = (xv[e] - mu) * rstd * g[c + e] + be[c + e];
    outF[(size_t)row * 1024 + c + e] = y;
    if (outB) outB[(size_t)row * 1024 + c + e] = f2bf(y);
  }
}

// ---------------------------------------------------------------- launcher
extern "C" void kernel_launch(void* const* d_in, const int* in_sizes, int n_in,
                              void* d_out, int out_size, void* d_ws, size_t ws_size,
                              hipStream_t stream) {
  const float* x    = (const float*)d_in[0];
  const int*   mask = (const int*)d_in[1];
  const float* Wq = (const float*)d_in[2];
  const float* bq = (const float*)d_in[3];
  const float* Wk = (const float*)d_in[4];
  const float* bk = (const float*)d_in[5];
  const float* Wv = (const float*)d_in[6];
  const float* bv = (const float*)d_in[7];
  const float* Wo = (const float*)d_in[8];
  const float* bo = (const float*)d_in[9];
  const float* ln1g = (const float*)d_in[10];
  const float* ln1b = (const float*)d_in[11];
  const float* ln2g = (const float*)d_in[12];
  const float* ln2b = (const float*)d_in[13];
  const float* W1 = (const float*)d_in[14];
  const float* b1 = (const float*)d_in[15];
  const float* W2 = (const float*)d_in[16];
  const float* b2 = (const float*)d_in[17];
  float* out = (float*)d_out;

  char* w = (char*)d_ws;
  auto alloc = [&](size_t bytes) {
    char* p = w;
    w += (bytes + 255) & ~(size_t)255;
    return p;
  };
  const size_t T = 4096;  // B*S tokens
  u16* xb    = (u16*)alloc(T * 1024 * 2);
  u16* wqcm  = (u16*)alloc(1024 * 1024 * 2);
  u16* wkcm  = (u16*)alloc(1024 * 1024 * 2);
  u16* wvcm  = (u16*)alloc(1024 * 1024 * 2);
  u16* wocm  = (u16*)alloc(1024 * 1024 * 2);
  u16* w1cm  = (u16*)alloc(4096 * 1024 * 2);
  u16* w2cm  = (u16*)alloc((size_t)1024 * 4096 * 2);
  u16* Qb    = (u16*)alloc(T * 1024 * 2);
  u16* Kb    = (u16*)alloc(T * 1024 * 2);
  u16* Vb    = (u16*)alloc(T * 1024 * 2);
  u16* ctx   = (u16*)alloc(T * 1024 * 2);
  float* s1  = (float*)alloc(T * 1024 * 4);
  float* hbf = (float*)alloc(T * 1024 * 4);
  u16* hb    = (u16*)alloc(T * 1024 * 2);
  u16* ff1b  = (u16*)alloc(T * 4096 * 2);
  float* s2  = (float*)alloc(T * 1024 * 4);

  // 1. convert & repack
  conv_bf16_k<<<16384, 256, 0, stream>>>(x, xb, 4194304);
  repack_w_k<<<4096, 256, 0, stream>>>(Wq, wqcm, 16, 1024, 64);
  repack_w_k<<<4096, 256, 0, stream>>>(Wk, wkcm, 16, 1024, 64);
  repack_w_k<<<4096, 256, 0, stream>>>(Wv, wvcm, 16, 1024, 64);
  repack_w_k<<<4096, 256, 0, stream>>>(Wo, wocm, 1, 1024, 1024);
  repack_w_k<<<16384, 256, 0, stream>>>(W1, w1cm, 1, 1024, 4096);
  repack_w_k<<<16384, 256, 0, stream>>>(W2, w2cm, 1, 4096, 1024);

  // 2. QKV projections (bf16 head layout)
  dim3 g1(16, 32);
  gemm_k<0><<<g1, 256, 0, stream>>>(xb, wqcm, bq, nullptr, nullptr, Qb, 4096, 1024, 1024);
  gemm_k<0><<<g1, 256, 0, stream>>>(xb, wkcm, bk, nullptr, nullptr, Kb, 4096, 1024, 1024);
  gemm_k<0><<<g1, 256, 0, stream>>>(xb, wvcm, bv, nullptr, nullptr, Vb, 4096, 1024, 1024);

  // 3. attention (flash, online softmax)
  attn_k<<<1024, 128, 0, stream>>>(Qb, Kb, Vb, mask, ctx);

  // 4. output projection + residual, then LN1
  gemm_k<1><<<g1, 256, 0, stream>>>(ctx, wocm, bo, x, s1, nullptr, 4096, 1024, 1024);
  ln_k<<<4096, 256, 0, stream>>>(s1, ln1g, ln1b, hbf, hb);

  // 5. FFN
  dim3 g2(64, 32);
  gemm_k<2><<<g2, 256, 0, stream>>>(hb, w1cm, b1, nullptr, nullptr, ff1b, 4096, 4096, 1024);
  gemm_k<1><<<g1, 256, 0, stream>>>(ff1b, w2cm, b2, hbf, s2, nullptr, 4096, 1024, 4096);

  // 6. LN2 -> output
  ln_k<<<4096, 256, 0, stream>>>(s2, ln2g, ln2b, out, nullptr);
}
